// SlotAttention_66039417143599
// MI455X (gfx1250) — compile-verified, matrix operand order checked by reading, not run on hardware
//
#include <hip/hip_runtime.h>
#include <hip/hip_bf16.h>

typedef __bf16 bf16_t;
typedef bf16_t v16bf __attribute__((ext_vector_type(16)));
typedef float  v8f   __attribute__((ext_vector_type(8)));
typedef unsigned int u32x4 __attribute__((ext_vector_type(4)));
typedef int          i32x8 __attribute__((ext_vector_type(8)));
typedef int          i32x4 __attribute__((ext_vector_type(4)));

union Frag32B { v16bf v; uint4 u[2]; };
union Pack16B { bf16_t h[8]; uint4 u; };

#define F_RELU   1
#define F_ACC    2
#define F_PADG   4
#define F_TRANS  8
#define F_STAGE  16   // stage B tile in LDS via TDM (requires K==256-style fit)
#define F_KSPLIT 32   // grid.x = K-split; write fp32 partials

#define LDS_PAD_ELEMS 8            // 16B pad per row -> bank-conflict-free frags

// ---------------------------------------------------------------------------
// Core bf16 WMMA tile op: D = A(16x32) * B(32x16) + C, fp32 accumulate.
// ---------------------------------------------------------------------------
__device__ __forceinline__ v8f wmma_bf16(const Frag32B& a, const Frag32B& b, v8f c) {
  return __builtin_amdgcn_wmma_f32_16x16x32_bf16(false, a.v, false, b.v,
                                                 (short)0, c, false, false);
}

// Per-lane fragment load for row-major [rows, K] bf16 (A operand, or B operand
// stored as B^T = [N, K] row-major):
//   lanes 0-15 : row = lane,    K chunks {k..k+7, k+16..k+23}
//   lanes 16-31: row = lane-16, K chunks {k+8..k+15, k+24..k+31}
__device__ __forceinline__ void load_frag(Frag32B& f, const bf16_t* rowp, int k0) {
  f.u[0] = *(const uint4*)(rowp + k0);
  f.u[1] = *(const uint4*)(rowp + k0 + 16);
}

// ---------------------------------------------------------------------------
// Tensor Data Mover: DMA a [rows x cols] bf16 tile (row stride srcStride elems)
// from global into LDS, optionally padding each 512B of data with 16B so LDS
// rows are skewed for conflict-free 16B fragment reads.
// D# built per CDNA5 ISA ch.8: group0 {count,lds,global,type}, group1 {dims}.
// Toolchain uses the 6-arg builtin: (g0, g1, g2, g3, g4, cpol).
// ---------------------------------------------------------------------------
__device__ __forceinline__ void tdm_load_tile(bf16_t* lds_dst, const bf16_t* gsrc,
                                              int rows, int cols, int srcStride,
                                              bool pad)
{
  const unsigned long long ga = (unsigned long long)(uintptr_t)gsrc;
  u32x4 g0;
  g0[0] = 1u;                                              // count=1, user mode
  g0[1] = (unsigned)(uintptr_t)lds_dst;                    // LDS byte address
  g0[2] = (unsigned)ga;                                    // global_addr[31:0]
  g0[3] = (unsigned)((ga >> 32) & 0x1FFFFFFu) | (2u << 30);// addr[56:32] | type=2

  int padbits = 0;
  if (pad) {
    // pad_interval code: interval = cols/2 dwords -> code = log2(cols/2) - 1
    const int icode = (31 - __builtin_clz((unsigned)(cols >> 1))) - 1;
    padbits = (1 << 20) | (icode << 22) | ((LDS_PAD_ELEMS / 2 - 1) << 25);
  }
  i32x8 g1;
  g1[0] = (1 << 16) | padbits;                             // data_size=2B
  g1[1] = (int)(((unsigned)cols & 0xFFFFu) << 16);         // tensor_dim0 lo
  g1[2] = (int)((((unsigned)cols >> 16) & 0xFFFFu) |
                (((unsigned)rows & 0xFFFFu) << 16));       // dim0 hi | dim1 lo
  g1[3] = (int)((((unsigned)rows >> 16) & 0xFFFFu) |
                (((unsigned)cols & 0xFFFFu) << 16));       // dim1 hi | tile_dim0
  g1[4] = rows & 0xFFFF;                                   // tile_dim1 (dim2=0)
  g1[5] = srcStride;                                       // dim0_stride lo32
  g1[6] = 0;
  g1[7] = 0;
  const i32x4 z4 = {0, 0, 0, 0};                           // groups 2/3 unused (2D)
  const i32x8 z8 = {0, 0, 0, 0, 0, 0, 0, 0};               // extra group (unused)
  __builtin_amdgcn_tensor_load_to_lds(g0, g1, z4, z4, z8, 0);
}

__device__ __forceinline__ void mma_loop(const bf16_t* __restrict__ arow,
                                         const bf16_t* __restrict__ b0,
                                         const bf16_t* __restrict__ b1,
                                         const bf16_t* __restrict__ b2,
                                         const bf16_t* __restrict__ b3,
                                         int kBegin, int kEnd, int hi, v8f acc[4])
{
  for (int kb = kBegin; kb < kEnd; kb += 32) {
    const int k0 = kb + hi * 8;
    Frag32B fa, fb;
    load_frag(fa, arow, k0);
    load_frag(fb, b0, k0); acc[0] = wmma_bf16(fa, fb, acc[0]);
    load_frag(fb, b1, k0); acc[1] = wmma_bf16(fa, fb, acc[1]);
    load_frag(fb, b2, k0); acc[2] = wmma_bf16(fa, fb, acc[2]);
    load_frag(fb, b3, k0); acc[3] = wmma_bf16(fa, fb, acc[3]);
  }
}

// ---------------------------------------------------------------------------
// Generic batched bf16 GEMM:  C[b] = act( A[b] (MxK) * Bm[b]^T (KxN) )
// Bm stored row-major [N, K].  Wave tile: 16 rows x 64 cols.
// F_STAGE : block's 64 B-rows are DMA'd to LDS once (TDM) and reused.
// F_KSPLIT: blockIdx.x indexes a K-chunk; fp32 partials written to Cf.
// ---------------------------------------------------------------------------
extern __shared__ char smem_raw[];

__global__ void gemm_bf16_wmma(
    const bf16_t* __restrict__ A, int lda, long long strideA,
    const bf16_t* __restrict__ Bm, int ldb, long long strideB,
    int M, int N, int K,
    float* __restrict__ Cf, int ldc, long long strideCf,
    bf16_t* __restrict__ Cb, long long strideCb,
    const float* __restrict__ bias,
    const float* __restrict__ rowscale, long long rsStride,
    float alpha, int flags, int ldT, int wcols)
{
  const int wave = threadIdx.x >> 5;
  const int lane = threadIdx.x & 31;
  const int wrow = wave / wcols;
  const int wcol = wave - wrow * wcols;
  const int nrw  = (blockDim.x >> 5) / wcols;
  const int c  = lane & 15;
  const int hi = lane >> 4;
  const int b = blockIdx.z;
  A  += (long long)b * strideA;
  Bm += (long long)b * strideB;

  int rowBase, kBegin, kEnd;
  if (flags & F_KSPLIT) {
    rowBase = wrow * 16;
    const int kchunk = K / gridDim.x;
    kBegin = blockIdx.x * kchunk;
    kEnd   = kBegin + kchunk;
  } else {
    rowBase = (blockIdx.x * nrw + wrow) * 16;
    kBegin = 0; kEnd = K;
  }
  const int colBase = (blockIdx.y * wcols + wcol) * 64;

  // --- optional TDM staging of this block's B tile ([wcols*64, K] rows) ----
  bf16_t* sB = (bf16_t*)smem_raw;
  const int ldsStride = K + LDS_PAD_ELEMS;
  if (flags & F_STAGE) {
    if (wave == 0) {
      tdm_load_tile(sB, Bm + (long long)(blockIdx.y * wcols * 64) * ldb,
                    wcols * 64, K, ldb, true);
      __builtin_amdgcn_s_wait_tensorcnt(0);
    }
    __syncthreads();
  }
  if (rowBase >= M || colBase >= N) return;

  const v8f zero = {0.f,0.f,0.f,0.f,0.f,0.f,0.f,0.f};
  v8f acc[4];
#pragma unroll
  for (int j = 0; j < 4; ++j) acc[j] = zero;

  const bf16_t* arow = A + (long long)(rowBase + c) * lda;

  if (flags & F_STAGE) {
    const bf16_t* b0 = sB + (long long)(wcol * 64 + c) * ldsStride;
    mma_loop(arow, b0, b0 + 16 * ldsStride, b0 + 32 * ldsStride,
             b0 + 48 * ldsStride, kBegin, kEnd, hi, acc);
  } else {
    const bf16_t* b0 = Bm + (long long)(colBase + c) * ldb;
    mma_loop(arow, b0, b0 + 16 * (long long)ldb, b0 + 32 * (long long)ldb,
             b0 + 48 * (long long)ldb, kBegin, kEnd, hi, acc);
  }

  // --- epilogue ------------------------------------------------------------
  if (flags & F_KSPLIT) {
    const long long base = (long long)(blockIdx.x * gridDim.z + b) * strideCf;
#pragma unroll
    for (int j = 0; j < 4; ++j) {
      const int col = colBase + j * 16 + c;
#pragma unroll
      for (int rr = 0; rr < 8; ++rr) {
        const int row = rowBase + rr + 8 * hi;
        Cf[base + (long long)row * ldc + col] = acc[j][rr];
      }
    }
    return;
  }

#pragma unroll
  for (int j = 0; j < 4; ++j) {
    const int col = colBase + j * 16 + c;
    const float bval = bias ? bias[col] : 0.f;
    if (flags & F_TRANS) {
      Pack16B pk;
#pragma unroll
      for (int rr = 0; rr < 8; ++rr) pk.h[rr] = (bf16_t)(acc[j][rr] * alpha);
      const long long grow = rowBase;                 // global row = bb*ldT + l
      const long long bb = grow / ldT;
      const long long l0 = (grow % ldT) + 8 * hi;
      *(uint4*)(Cb + (bb * (long long)N + col) * (long long)ldT + l0) = pk.u;
      continue;
    }
#pragma unroll
    for (int rr = 0; rr < 8; ++rr) {
      const int row = rowBase + rr + 8 * hi;
      float val = acc[j][rr] * alpha + bval;
      if (flags & F_RELU) val = fmaxf(val, 0.f);
      if (rowscale) val *= rowscale[(long long)b * rsStride + row];
      if ((flags & F_PADG) && ((row & 15) >= 11)) continue;
      if (Cf) {
        float* p = Cf + (long long)b * strideCf + (long long)row * ldc + col;
        if (flags & F_ACC) *p = *p + val; else *p = val;
      }
      if (Cb)
        Cb[(long long)b * strideCb + (long long)row * ldc + col] = (bf16_t)val;
    }
  }
}

// Deterministic reduction of K-split partials; folds in per-row scale.
__global__ void ksplit_reduce_kernel(const float* __restrict__ part, // [ns,B,16,256]
                                     int nsplit,
                                     const float* __restrict__ rowscale, // [B*16]
                                     bf16_t* __restrict__ out)           // [B*16,256]
{
  const long long i = (long long)blockIdx.x * blockDim.x + threadIdx.x;
  const int col = (int)(i & 255);
  const long long row = i >> 8;               // b*16 + o
  float s = 0.f;
  for (int sp = 0; sp < nsplit; ++sp)
    s += part[((long long)sp * 1024 + row) * 256 + col];
  out[i] = (bf16_t)(s * rowscale[row]);
}

// ---------------------------------------------------------------------------
// LayerNorm over last dim (256) -> bf16.  One wave per row, 8 elems/lane.
// ---------------------------------------------------------------------------
__global__ void ln_rows_kernel(const float* __restrict__ X,
                               const float* __restrict__ g,
                               const float* __restrict__ bta,
                               bf16_t* __restrict__ Y,
                               long long nrows)
{
  const long long row = (long long)blockIdx.x * (blockDim.x >> 5) + (threadIdx.x >> 5);
  if (row >= nrows) return;
  const int lane = threadIdx.x & 31;
  const float* xr = X + row * 256 + lane * 8;
  const float4 a  = *(const float4*)(xr);
  const float4 bq = *(const float4*)(xr + 4);
  float v[8] = {a.x, a.y, a.z, a.w, bq.x, bq.y, bq.z, bq.w};
  float s = 0.f;
#pragma unroll
  for (int i = 0; i < 8; ++i) s += v[i];
#pragma unroll
  for (int off = 16; off > 0; off >>= 1) s += __shfl_xor(s, off, 32);
  const float mu = s * (1.0f / 256.0f);
  float vs = 0.f;
#pragma unroll
  for (int i = 0; i < 8; ++i) { v[i] -= mu; vs += v[i] * v[i]; }
#pragma unroll
  for (int off = 16; off > 0; off >>= 1) vs += __shfl_xor(vs, off, 32);
  const float rinv = rsqrtf(vs * (1.0f / 256.0f) + 1e-6f);
  const float4 g0 = *(const float4*)(g + lane * 8);
  const float4 g1 = *(const float4*)(g + lane * 8 + 4);
  const float4 t0 = *(const float4*)(bta + lane * 8);
  const float4 t1 = *(const float4*)(bta + lane * 8 + 4);
  const float gg[8] = {g0.x, g0.y, g0.z, g0.w, g1.x, g1.y, g1.z, g1.w};
  const float bb[8] = {t0.x, t0.y, t0.z, t0.w, t1.x, t1.y, t1.z, t1.w};
  Pack16B pk;
#pragma unroll
  for (int i = 0; i < 8; ++i) pk.h[i] = (bf16_t)(v[i] * rinv * gg[i] + bb[i]);
  *(uint4*)(Y + row * 256 + lane * 8) = pk.u;
}

// ---------------------------------------------------------------------------
// Fused logits + inverted softmax.  q tile (16x256) staged in LDS via TDM.
// WMMA gives a 16(slot) x 16(L) tile; softmax runs down each tile column,
// held by lane pair (j, j+16).
// ---------------------------------------------------------------------------
__global__ void qk_softmax_kernel(const bf16_t* __restrict__ q,   // [B*16,256]
                                  const bf16_t* __restrict__ k,   // [B*L,256]
                                  float* __restrict__ attn_out,   // [B,11,L]
                                  bf16_t* __restrict__ attn_bf,   // [B,16,L]
                                  int L)
{
  __shared__ bf16_t sQ[16 * (256 + LDS_PAD_ELEMS)];
  const int b = blockIdx.y;
  const int lane = threadIdx.x & 31;
  const int wave = threadIdx.x >> 5;
  if (wave == 0) {
    tdm_load_tile(sQ, q + (long long)b * 16 * 256, 16, 256, 256, true);
    __builtin_amdgcn_s_wait_tensorcnt(0);
  }
  __syncthreads();

  const int l0 = (blockIdx.x * (blockDim.x >> 5) + wave) * 16;
  if (l0 >= L) return;
  const int c = lane & 15, hi = lane >> 4;
  const bf16_t* arow = sQ + c * (256 + LDS_PAD_ELEMS);
  const bf16_t* brow = k + ((long long)b * L + l0 + c) * 256;

  v8f acc = {0.f,0.f,0.f,0.f,0.f,0.f,0.f,0.f};
#pragma unroll
  for (int kb = 0; kb < 256; kb += 32) {
    const int k0 = kb + hi * 8;
    Frag32B fa, fb;
    load_frag(fa, arow, k0);
    load_frag(fb, brow, k0);
    acc = wmma_bf16(fa, fb, acc);
  }

  float vals[8];
  float m = -3.0e38f;
#pragma unroll
  for (int rr = 0; rr < 8; ++rr) {
    const int row = rr + 8 * hi;
    const float v = (row < 11) ? acc[rr] : -3.0e38f;
    vals[rr] = v;
    m = fmaxf(m, v);
  }
  m = fmaxf(m, __shfl_xor(m, 16, 32));
  float s = 0.f;
#pragma unroll
  for (int rr = 0; rr < 8; ++rr) {
    const int row = rr + 8 * hi;
    const float e = (row < 11) ? expf(vals[rr] - m) : 0.f;
    vals[rr] = e;
    s += e;
  }
  s += __shfl_xor(s, 16, 32);
  const float inv = 1.0f / s;
#pragma unroll
  for (int rr = 0; rr < 8; ++rr) {
    const int row = rr + 8 * hi;
    const float a = vals[rr] * inv;
    attn_bf[((long long)b * 16 + row) * L + l0 + c] = (bf16_t)a;   // pads -> 0
    if (row < 11)
      attn_out[((long long)b * 11 + row) * L + l0 + c] = a;
  }
}

// Per-(b,o) reciprocal of attn row-sum over L (deterministic LDS reduction).
__global__ void rowsum_recip_kernel(const float* __restrict__ attn,  // [B,11,L]
                                    float* __restrict__ recip,       // [B*16]
                                    int L)
{
  const int row = blockIdx.x;        // b*16 + o
  const int o = row & 15, b = row >> 4;
  __shared__ float red[256];
  float s = 0.f;
  if (o < 11) {
    const float* ar = attn + ((long long)b * 11 + o) * L;
    for (int i = threadIdx.x; i < L; i += blockDim.x) s += ar[i];
  }
  red[threadIdx.x] = s;
  __syncthreads();
  for (int st = 128; st > 0; st >>= 1) {
    if ((int)threadIdx.x < st) red[threadIdx.x] += red[threadIdx.x + st];
    __syncthreads();
  }
  if (threadIdx.x == 0) recip[row] = (o < 11) ? (1.0f / (red[0] + 1e-8f)) : 0.0f;
}

// GRU cell elementwise update (pads skipped; slots stays zero there).
__global__ void gru_kernel(const float* __restrict__ gi,   // [1024,768]
                           const float* __restrict__ gh,   // [1024,768]
                           float* __restrict__ slots)      // [1024,256]
{
  const long long i = (long long)blockIdx.x * blockDim.x + threadIdx.x;
  const int col = (int)(i & 255);
  const int row = (int)(i >> 8);
  if ((row & 15) >= 11) return;
  const float* gir = gi + (long long)row * 768;
  const float* ghr = gh + (long long)row * 768;
  const float ir = gir[col], iz = gir[col + 256], in = gir[col + 512];
  const float hr = ghr[col], hz = ghr[col + 256], hn = ghr[col + 512];
  const float r = 1.0f / (1.0f + expf(-(ir + hr)));
  const float z = 1.0f / (1.0f + expf(-(iz + hz)));
  const float n = tanhf(in + r * hn);
  slots[i] = (1.0f - z) * n + z * slots[i];
}

__global__ void cast_bf16_kernel(const float* __restrict__ x,
                                 bf16_t* __restrict__ y, long long n)
{
  const long long i = (long long)blockIdx.x * blockDim.x + threadIdx.x;
  if (i < n) y[i] = (bf16_t)x[i];
}

// d_in slots [B,11,256] -> padded fp32 [B,16,256] (pads zeroed)
__global__ void init_slots_kernel(const float* __restrict__ sin,
                                  float* __restrict__ sws)
{
  const long long i = (long long)blockIdx.x * blockDim.x + threadIdx.x; // B*16*256
  const int col = (int)(i & 255);
  const long long row = i >> 8;
  const int o = (int)(row & 15);
  const long long b = row >> 4;
  sws[i] = (o < 11) ? sin[(b * 11 + o) * 256 + col] : 0.0f;
}

// padded fp32 [B,16,256] -> d_out slots [B,11,256]
__global__ void copy_slots_out_kernel(const float* __restrict__ sws,
                                      float* __restrict__ out)
{
  const long long i = (long long)blockIdx.x * blockDim.x + threadIdx.x; // B*11*256
  const int col = (int)(i & 255);
  const long long row = i >> 8;           // b*11 + o
  const long long b = row / 11;
  const int o = (int)(row - b * 11);
  out[i] = sws[((b * 16 + o) << 8) + col];
}

// ---------------------------------------------------------------------------
// Host orchestration
// ---------------------------------------------------------------------------
static void launch_gemm(hipStream_t st,
                        const bf16_t* A, int lda, long long sA,
                        const bf16_t* Bm, int ldb, long long sB,
                        int M, int N, int K,
                        float* Cf, int ldc, long long sCf,
                        bf16_t* Cb, long long sCb,
                        const float* bias, const float* rowscale, long long rsStride,
                        float alpha, int flags, int ldT,
                        int wrows, int wcols, int batch, int ksplit = 1)
{
  dim3 grid((flags & F_KSPLIT) ? ksplit : (M + wrows * 16 - 1) / (wrows * 16),
            (N + wcols * 64 - 1) / (wcols * 64), batch);
  dim3 block(32 * wrows * wcols);
  size_t shmem = (flags & F_STAGE)
                     ? (size_t)(wcols * 64) * (K + LDS_PAD_ELEMS) * sizeof(bf16_t)
                     : 0;
  gemm_bf16_wmma<<<grid, block, shmem, st>>>(A, lda, sA, Bm, ldb, sB, M, N, K,
                                             Cf, ldc, sCf, Cb, sCb, bias,
                                             rowscale, rsStride, alpha, flags,
                                             ldT, wcols);
}

extern "C" void kernel_launch(void* const* d_in, const int* in_sizes, int n_in,
                              void* d_out, int out_size, void* d_ws, size_t ws_size,
                              hipStream_t stream) {
  (void)in_sizes; (void)n_in; (void)out_size; (void)ws_size;
  constexpr int  B = 64, O = 11, L = 4096, S = 256, H = 1024, OP = 16;
  constexpr long long BL = (long long)B * L;     // 262144
  constexpr int  R = B * OP;                      // 1024 padded slot rows
  constexpr int  NSPLIT = 8;                      // K-split for updates GEMM

  const float* in_slots = (const float*)d_in[0];
  const float* in_x     = (const float*)d_in[1];
  const float* ln_in_g  = (const float*)d_in[2];
  const float* ln_in_b  = (const float*)d_in[3];
  const float* Wk       = (const float*)d_in[4];
  const float* Wv       = (const float*)d_in[5];
  const float* ln_q_g   = (const float*)d_in[6];
  const float* ln_q_b   = (const float*)d_in[7];
  const float* Wq       = (const float*)d_in[8];
  const float* gru_wi   = (const float*)d_in[9];
  const float* gru_wh   = (const float*)d_in[10];
  const float* gru_bi   = (const float*)d_in[11];
  const float* gru_bh   = (const float*)d_in[12];
  const float* ln_m_g   = (const float*)d_in[13];
  const float* ln_m_b   = (const float*)d_in[14];
  const float* W1       = (const float*)d_in[15];
  const float* b1       = (const float*)d_in[16];
  const float* W2       = (const float*)d_in[17];
  const float* b2       = (const float*)d_in[18];

  float* out_slots = (float*)d_out;                         // [B,11,256]
  float* out_attn  = (float*)d_out + (long long)B * O * S;  // [B,11,L]

  // ---- workspace carve-up -------------------------------------------------
  char* ws = (char*)d_ws;
  size_t off = 0;
  auto alloc = [&](size_t bytes) -> char* {
    char* p = ws + off;
    off += (bytes + 255) & ~(size_t)255;
    return p;
  };
  bf16_t* x_bf   = (bf16_t*)alloc(BL * S * 2);          // LN(inputs)
  bf16_t* k_bf   = (bf16_t*)alloc(BL * S * 2);          // [B*L, S]
  bf16_t* vT_bf  = (bf16_t*)alloc(BL * S * 2);          // [B, S, L]
  bf16_t* wk_bf  = (bf16_t*)alloc((size_t)S * S * 2);
  bf16_t* wv_bf  = (bf16_t*)alloc((size_t)S * S * 2);
  bf16_t* wq_bf  = (bf16_t*)alloc((size_t)S * S * 2);
  bf16_t* wi_bf  = (bf16_t*)alloc((size_t)3 * S * S * 2);
  bf16_t* wh_bf  = (bf16_t*)alloc((size_t)3 * S * S * 2);
  bf16_t* w1_bf  = (bf16_t*)alloc((size_t)H * S * 2);
  bf16_t* w2_bf  = (bf16_t*)alloc((size_t)S * H * 2);
  float*  slots_ws = (float*)alloc((size_t)R * S * 4);  // padded fp32 slots
  bf16_t* sn_bf    = (bf16_t*)alloc((size_t)R * S * 2); // LN(slots)
  bf16_t* slots_bf = (bf16_t*)alloc((size_t)R * S * 2); // raw cast
  bf16_t* q_bf     = (bf16_t*)alloc((size_t)R * S * 2);
  bf16_t* attn_bf  = (bf16_t*)alloc((size_t)B * OP * L * 2);
  float*  recip    = (float*)alloc((size_t)R * 4);
  bf16_t* upd_bf   = (bf16_t*)alloc((size_t)R * S * 2);
  float*  part_f   = (float*)alloc((size_t)NSPLIT * R * S * 4);
  float*  gi_f     = (float*)alloc((size_t)R * 3 * S * 4);
  float*  gh_f     = (float*)alloc((size_t)R * 3 * S * 4);
  bf16_t* h_bf     = (bf16_t*)alloc((size_t)R * S * 2);
  bf16_t* t_bf     = (bf16_t*)alloc((size_t)R * H * 2);

  auto cast = [&](const float* src, bf16_t* dst, long long n) {
    cast_bf16_kernel<<<dim3((unsigned)((n + 255) / 256)), dim3(256), 0, stream>>>(src, dst, n);
  };

  // ---- one-time (per call) preprocessing ----------------------------------
  cast(Wk, wk_bf, (long long)S * S);
  cast(Wv, wv_bf, (long long)S * S);
  cast(Wq, wq_bf, (long long)S * S);
  cast(gru_wi, wi_bf, (long long)3 * S * S);
  cast(gru_wh, wh_bf, (long long)3 * S * S);
  cast(W1, w1_bf, (long long)H * S);
  cast(W2, w2_bf, (long long)S * H);

  ln_rows_kernel<<<dim3((unsigned)(BL / 8)), dim3(256), 0, stream>>>(
      in_x, ln_in_g, ln_in_b, x_bf, BL);

  // k = x @ Wk^T  -> bf16 [B*L, S]            (B tile TDM-staged in LDS)
  launch_gemm(stream, x_bf, S, 0, wk_bf, S, 0, (int)BL, S, S,
              nullptr, S, 0, k_bf, 0, nullptr, nullptr, 0,
              1.0f, F_STAGE, 0, /*wrows*/8, /*wcols*/1, 1);
  // v = x @ Wv^T  -> bf16 transposed [B, S, L]
  launch_gemm(stream, x_bf, S, 0, wv_bf, S, 0, (int)BL, S, S,
              nullptr, S, 0, vT_bf, 0, nullptr, nullptr, 0,
              1.0f, F_TRANS | F_STAGE, L, 8, 1, 1);

  init_slots_kernel<<<dim3((unsigned)(R * S / 256)), dim3(256), 0, stream>>>(
      in_slots, slots_ws);

  const float scale = 0.0625f;   // S^-0.5 = 1/16

  for (int it = 0; it < 2; ++it) {
    // sn = LN(slots); raw bf16 copy of slots for the GRU hidden path
    ln_rows_kernel<<<dim3(R / 8), dim3(256), 0, stream>>>(
        slots_ws, ln_q_g, ln_q_b, sn_bf, R);
    cast(slots_ws, slots_bf, (long long)R * S);

    // q = (sn @ Wq^T) * scale   [R, S] bf16
    launch_gemm(stream, sn_bf, S, 0, wq_bf, S, 0, R, S, S,
                nullptr, S, 0, q_bf, 0, nullptr, nullptr, 0,
                scale, F_STAGE, 0, 8, 1, 1);

    // logits + inverted softmax -> attn (fp32 to d_out) + attn_bf (padded)
    qk_softmax_kernel<<<dim3(L / 128, B), dim3(256), 0, stream>>>(
        q_bf, k_bf, out_attn, attn_bf, L);

    // per-(b,o) 1/(sum_L attn + eps)
    rowsum_recip_kernel<<<dim3(R), dim3(256), 0, stream>>>(out_attn, recip, L);

    // updates = (attn / rowsum) @ v : batched, K split 8 ways for parallelism
    launch_gemm(stream, attn_bf, L, (long long)OP * L,
                vT_bf, L, (long long)S * L, OP, S, L,
                part_f, S, (long long)OP * S, nullptr, 0,
                nullptr, nullptr, 0, 1.0f, F_KSPLIT, 0, 1, 4, B, NSPLIT);
    ksplit_reduce_kernel<<<dim3(R * S / 256), dim3(256), 0, stream>>>(
        part_f, NSPLIT, recip, upd_bf);

    // GRU gates
    launch_gemm(stream, upd_bf, S, 0, wi_bf, S, 0, R, 3 * S, S,
                gi_f, 3 * S, 0, nullptr, 0, gru_bi, nullptr, 0,
                1.0f, F_STAGE, 0, 8, 1, 1);
    launch_gemm(stream, slots_bf, S, 0, wh_bf, S, 0, R, 3 * S, S,
                gh_f, 3 * S, 0, nullptr, 0, gru_bh, nullptr, 0,
                1.0f, F_STAGE, 0, 8, 1, 1);
    gru_kernel<<<dim3(R * S / 256), dim3(256), 0, stream>>>(gi_f, gh_f, slots_ws);

    // pre-LN residual MLP
    ln_rows_kernel<<<dim3(R / 8), dim3(256), 0, stream>>>(
        slots_ws, ln_m_g, ln_m_b, h_bf, R);
    launch_gemm(stream, h_bf, S, 0, w1_bf, S, 0, R, H, S,
                nullptr, H, 0, t_bf, 0, b1, nullptr, 0,
                1.0f, F_RELU | F_STAGE, 0, 8, 1, 1);
    launch_gemm(stream, t_bf, H, 0, w2_bf, H, 0, R, S, H,
                slots_ws, S, 0, nullptr, 0, b2, nullptr, 0,
                1.0f, F_ACC | F_PADG, 0, 4, 2, 1);
  }

  copy_slots_out_kernel<<<dim3((unsigned)((long long)B * O * S / 256)),
                          dim3(256), 0, stream>>>(slots_ws, out_slots);
}
